// MultiHeadAttention_1443109012141
// MI455X (gfx1250) — compile-verified, matrix operand order checked by reading, not run on hardware
//
#include <hip/hip_runtime.h>
#include <hip/hip_bf16.h>
#include <math.h>

typedef unsigned short u16;
typedef __attribute__((ext_vector_type(16))) __bf16 v16bf;
typedef __attribute__((ext_vector_type(8)))  float  v8f;

#define D_   1024
#define S_   2048
#define H_   16
#define HD_  64
#define MTOT 4096   // B*S

__device__ __forceinline__ u16 f2bf(float f) {
    unsigned int u = __float_as_uint(f);
    unsigned int r = u + 0x7FFFu + ((u >> 16) & 1u);   // round-to-nearest-even
    return (u16)(r >> 16);
}

__device__ __forceinline__ v8f vzero() {
    v8f z;
#pragma unroll
    for (int i = 0; i < 8; ++i) z[i] = 0.f;
    return z;
}

// 16x32 bf16 WMMA A-operand fragment (also B-operand when the source is
// stored N-major, i.e. B^T row-major). Layout per ISA 7.12.2.
__device__ __forceinline__ v16bf load_frag(const u16* base, int ld, int row0, int k0) {
    int lane = threadIdx.x & 31;
    int half = lane >> 4;
    int r    = row0 + (lane & 15);
    const u16* p = base + (long)r * ld + k0 + half * 8;
    uint4 lo = *(const uint4*)p;          // 8 bf16
    uint4 hi = *(const uint4*)(p + 16);   // 8 bf16
    v16bf f;
    ((uint4*)&f)[0] = lo;
    ((uint4*)&f)[1] = hi;
    return f;
}

__device__ __forceinline__ v8f wmma_bf16(v16bf a, v16bf b, v8f c) {
    return __builtin_amdgcn_wmma_f32_16x16x32_bf16(false, a, false, b, (short)0, c, false, false);
}

// Async DMA of 16 bytes global -> LDS (ASYNCcnt-tracked, no VGPR staging).
// VDST = 32-bit LDS byte address (low 32 bits of the flat pointer, per the
// LDS aperture rule), VADDR = 64-bit global address, SADDR = off.
__device__ __forceinline__ void async_cp16(const u16* g, const u16* l) {
    unsigned lds = (unsigned)(unsigned long long)l;
    asm volatile("global_load_async_to_lds_b128 %0, %1, off"
                 :: "v"(lds), "v"(g) : "memory");
}

// ---------------------------------------------------------------- fp32->bf16
__global__ __launch_bounds__(256) void mha_cvt_bf16(const float* __restrict__ src,
                                                    u16* __restrict__ dst, int n) {
    int i = (blockIdx.x * 256 + threadIdx.x) * 4;
    if (i + 3 >= n) return;
    float4 v = *(const float4*)(src + i);
    unsigned long long pk =
        (unsigned long long)f2bf(v.x)        |
        ((unsigned long long)f2bf(v.y) << 16)|
        ((unsigned long long)f2bf(v.z) << 32)|
        ((unsigned long long)f2bf(v.w) << 48);
    *(unsigned long long*)(dst + i) = pk;
}

// ------------------------------------------------------- QKV projection GEMM
// Y = x @ W^T + b. Wave tile 32x64 (2x4 accumulators), WG tile 128x128.
// transposed==0 : out[b][h][s][hd]   (Q, K)
// transposed==1 : out[b][h][hd][s]   (V, pre-transposed for the PV WMMA)
__global__ __launch_bounds__(256) void mha_qkv_gemm(const u16* __restrict__ A,
                                                    const u16* __restrict__ W,
                                                    const float* __restrict__ bias,
                                                    u16* __restrict__ out,
                                                    int transposed) {
    int wave = threadIdx.x >> 5;
    int lane = threadIdx.x & 31;
    int half = lane >> 4;
    int mBase = blockIdx.x * 128 + (wave >> 1) * 32;
    int nBase = blockIdx.y * 128 + (wave & 1) * 64;

    v8f acc[2][4];
#pragma unroll
    for (int i = 0; i < 2; ++i)
#pragma unroll
        for (int j = 0; j < 4; ++j) acc[i][j] = vzero();

    for (int k0 = 0; k0 < D_; k0 += 32) {
        v16bf a0 = load_frag(A, D_, mBase,      k0);
        v16bf a1 = load_frag(A, D_, mBase + 16, k0);
#pragma unroll
        for (int ni = 0; ni < 4; ++ni) {
            v16bf bf = load_frag(W, D_, nBase + ni * 16, k0);  // W row n == col n of W^T
            acc[0][ni] = wmma_bf16(a0, bf, acc[0][ni]);
            acc[1][ni] = wmma_bf16(a1, bf, acc[1][ni]);
        }
    }

#pragma unroll
    for (int mi = 0; mi < 2; ++mi)
#pragma unroll
        for (int ni = 0; ni < 4; ++ni) {
            int col = nBase + ni * 16 + (lane & 15);
            float bia = bias[col];
            int h  = col >> 6;
            int hd = col & 63;
#pragma unroll
            for (int i = 0; i < 8; ++i) {
                int rowM = mBase + mi * 16 + i + half * 8;
                int b = rowM >> 11;          // / S_
                int s = rowM & (S_ - 1);
                u16 bf = f2bf(acc[mi][ni][i] + bia);
                if (!transposed)
                    out[(((long)(b * H_ + h) * S_ + s) * HD_) + hd] = bf;
                else
                    out[(((long)(b * H_ + h) * HD_ + hd) * S_) + s] = bf;
            }
        }
}

// ----------------------------------------------------- flash attention core
// 4 waves share one (b,h); K/V tiles are staged cooperatively into LDS with
// async DMA, double-buffered so the next tile's copy overlaps this tile's
// WMMA + online softmax. Each wave owns a 32-query tile.
__global__ __launch_bounds__(128) void mha_attn(const u16* __restrict__ Qh,
                                                const u16* __restrict__ Kh,
                                                const u16* __restrict__ Vt,
                                                const int* __restrict__ mask,
                                                u16* __restrict__ ctx) {
    __shared__ __align__(16) u16 Kt[2][32 * 64];    // 2 x 4KB
    __shared__ __align__(16) u16 Vl[2][64 * 32];    // 2 x 4KB
    __shared__ __align__(16) u16 Plds[4][32 * 32];  // 8KB

    int tid  = threadIdx.x;
    int wave = tid >> 5;
    int lane = tid & 31;
    int half = lane >> 4;
    int bh = blockIdx.x;               // 0..31
    int b  = bh >> 4, h = bh & 15;
    int q0 = (blockIdx.y * 4 + wave) * 32;

    const u16* Qb = Qh + (long)bh * S_ * HD_;   // [S][64]
    const u16* Kb = Kh + (long)bh * S_ * HD_;   // [S][64]
    const u16* Vb = Vt + (long)bh * HD_ * S_;   // [64][S]
    u16* P = Plds[wave];

    // Q fragments stay resident for the whole key loop.
    v16bf qf[2][2];
#pragma unroll
    for (int mi = 0; mi < 2; ++mi)
#pragma unroll
        for (int ks = 0; ks < 2; ++ks)
            qf[mi][ks] = load_frag(Qb, HD_, q0 + mi * 16, ks * 32);

    float mrow[2][8], lrow[2][8];
#pragma unroll
    for (int mi = 0; mi < 2; ++mi)
#pragma unroll
        for (int i = 0; i < 8; ++i) { mrow[mi][i] = -1e30f; lrow[mi][i] = 0.f; }

    v8f o[2][4];
#pragma unroll
    for (int mi = 0; mi < 2; ++mi)
#pragma unroll
        for (int nt = 0; nt < 4; ++nt) o[mi][nt] = vzero();

    // Cooperative async stage of one key tile (kt..kt+31) into buffer `bi`.
    // K tile: 32 rows x 64 elems, contiguous 4KB. V tile: 64 rows x 32 elems.
    // 128 threads x 2 chunks x 16B per matrix = 4 async ops per thread.
    auto prefetch = [&](int kt, int bi) {
#pragma unroll
        for (int j = 0; j < 2; ++j) {
            int c = tid + j * 128;                       // 0..255 16B chunks
            async_cp16(Kb + (long)kt * HD_ + c * 8, &Kt[bi][c * 8]);
            async_cp16(Vb + (long)(c >> 2) * S_ + kt + (c & 3) * 8, &Vl[bi][c * 8]);
        }
    };

    prefetch(0, 0);
    const int NKT = S_ / 32;
    for (int it = 0; it < NKT; ++it) {
        int kt = it * 32;
        int cur = it & 1;
        if (it + 1 < NKT) {
            prefetch(kt + 32, cur ^ 1);
            asm volatile("s_wait_asynccnt 0x4" ::: "memory");  // only next tile in flight
        } else {
            asm volatile("s_wait_asynccnt 0x0" ::: "memory");
        }
        __syncthreads();   // all waves' DMA portions have landed

        // ---- scores = Q . K^T from LDS
        v8f sc[2][2];
#pragma unroll
        for (int mi = 0; mi < 2; ++mi)
#pragma unroll
            for (int ni = 0; ni < 2; ++ni) sc[mi][ni] = vzero();
#pragma unroll
        for (int ni = 0; ni < 2; ++ni) {
            v16bf kf0 = load_frag(Kt[cur], HD_, ni * 16, 0);
            v16bf kf1 = load_frag(Kt[cur], HD_, ni * 16, 32);
#pragma unroll
            for (int mi = 0; mi < 2; ++mi) {
                sc[mi][ni] = wmma_bf16(qf[mi][0], kf0, sc[mi][ni]);
                sc[mi][ni] = wmma_bf16(qf[mi][1], kf1, sc[mi][ni]);
            }
        }

        // ---- scale + additive mask (column == key index lives per lane)
        float madd[2];
#pragma unroll
        for (int ni = 0; ni < 2; ++ni) {
            int key = kt + ni * 16 + (lane & 15);
            madd[ni] = (mask[b * S_ + key] == 0) ? -1e30f : 0.f;
        }
#pragma unroll
        for (int mi = 0; mi < 2; ++mi)
#pragma unroll
            for (int ni = 0; ni < 2; ++ni)
#pragma unroll
                for (int i = 0; i < 8; ++i)
                    sc[mi][ni][i] = sc[mi][ni][i] * 0.125f + madd[ni];

        // ---- online softmax; rows live across one 16-lane half per VGPR
#pragma unroll
        for (int mi = 0; mi < 2; ++mi)
#pragma unroll
            for (int i = 0; i < 8; ++i) {
                float v = fmaxf(sc[mi][0][i], sc[mi][1][i]);
#pragma unroll
                for (int d = 8; d >= 1; d >>= 1) v = fmaxf(v, __shfl_xor(v, d, 16));
                float mnew  = fmaxf(mrow[mi][i], v);
                float scale = __expf(mrow[mi][i] - mnew);
                mrow[mi][i] = mnew;
                lrow[mi][i] *= scale;
#pragma unroll
                for (int nt = 0; nt < 4; ++nt) o[mi][nt][i] *= scale;
                float p0 = __expf(sc[mi][0][i] - mnew);
                float p1 = __expf(sc[mi][1][i] - mnew);
                sc[mi][0][i] = p0;
                sc[mi][1][i] = p1;
                float s = p0 + p1;
#pragma unroll
                for (int d = 8; d >= 1; d >>= 1) s += __shfl_xor(s, d, 16);
                lrow[mi][i] += s;
            }

        // ---- P: C-layout f32 -> row-major bf16 in per-wave LDS tile
#pragma unroll
        for (int mi = 0; mi < 2; ++mi)
#pragma unroll
            for (int ni = 0; ni < 2; ++ni)
#pragma unroll
                for (int i = 0; i < 8; ++i)
                    P[(mi * 16 + i + half * 8) * 32 + ni * 16 + (lane & 15)] =
                        f2bf(sc[mi][ni][i]);
        __syncthreads();

        // ---- O += P . V  (V pre-transposed: B operand row-major over keys)
        v16bf pa0 = load_frag(P, 32, 0,  0);
        v16bf pa1 = load_frag(P, 32, 16, 0);
#pragma unroll
        for (int nt = 0; nt < 4; ++nt) {
            v16bf vf = load_frag(Vl[cur], 32, nt * 16, 0);
            o[0][nt] = wmma_bf16(pa0, vf, o[0][nt]);
            o[1][nt] = wmma_bf16(pa1, vf, o[1][nt]);
        }
        __syncthreads();   // buffer `cur` free for the it+2 prefetch
    }

    // ---- normalize and store ctx [B*S][D] bf16
#pragma unroll
    for (int mi = 0; mi < 2; ++mi)
#pragma unroll
        for (int i = 0; i < 8; ++i) {
            float inv = 1.f / lrow[mi][i];
            int srow = q0 + mi * 16 + i + half * 8;
            long gRow = (long)b * S_ + srow;
#pragma unroll
            for (int nt = 0; nt < 4; ++nt) {
                int col = h * HD_ + nt * 16 + (lane & 15);
                ctx[gRow * D_ + col] = f2bf(o[mi][nt][i] * inv);
            }
        }
}

// ------------------------------------------------------ output projection
__global__ __launch_bounds__(256) void mha_proj(const u16* __restrict__ A,
                                                const u16* __restrict__ W,
                                                const float* __restrict__ bias,
                                                float* __restrict__ out) {
    int wave = threadIdx.x >> 5;
    int lane = threadIdx.x & 31;
    int half = lane >> 4;
    int mBase = blockIdx.x * 128 + (wave >> 1) * 32;
    int nBase = blockIdx.y * 128 + (wave & 1) * 64;

    v8f acc[2][4];
#pragma unroll
    for (int i = 0; i < 2; ++i)
#pragma unroll
        for (int j = 0; j < 4; ++j) acc[i][j] = vzero();

    for (int k0 = 0; k0 < D_; k0 += 32) {
        v16bf a0 = load_frag(A, D_, mBase,      k0);
        v16bf a1 = load_frag(A, D_, mBase + 16, k0);
#pragma unroll
        for (int ni = 0; ni < 4; ++ni) {
            v16bf bf = load_frag(W, D_, nBase + ni * 16, k0);
            acc[0][ni] = wmma_bf16(a0, bf, acc[0][ni]);
            acc[1][ni] = wmma_bf16(a1, bf, acc[1][ni]);
        }
    }

#pragma unroll
    for (int mi = 0; mi < 2; ++mi)
#pragma unroll
        for (int ni = 0; ni < 4; ++ni) {
            int col = nBase + ni * 16 + (lane & 15);
            float bia = bias[col];
#pragma unroll
            for (int i = 0; i < 8; ++i) {
                int rowM = mBase + mi * 16 + i + half * 8;
                out[(long)rowM * D_ + col] = acc[mi][ni][i] + bia;
            }
        }
}

extern "C" void kernel_launch(void* const* d_in, const int* in_sizes, int n_in,
                              void* d_out, int out_size, void* d_ws, size_t ws_size,
                              hipStream_t stream) {
    const float* x    = (const float*)d_in[0];
    const int*   mask = (const int*)  d_in[1];
    const float* Wq   = (const float*)d_in[2];
    const float* bq   = (const float*)d_in[3];
    const float* Wk   = (const float*)d_in[4];
    const float* bk   = (const float*)d_in[5];
    const float* Wv   = (const float*)d_in[6];
    const float* bv   = (const float*)d_in[7];
    const float* Wo   = (const float*)d_in[8];
    const float* bo   = (const float*)d_in[9];

    const size_t MB = 1024 * 1024;
    char* ws = (char*)d_ws;
    u16* xb  = (u16*)(ws + 0 * MB);    // 8 MB  [4096][1024]
    u16* Wqb = (u16*)(ws + 8 * MB);    // 2 MB
    u16* Wkb = (u16*)(ws + 10 * MB);
    u16* Wvb = (u16*)(ws + 12 * MB);
    u16* Wob = (u16*)(ws + 14 * MB);
    u16* Qh  = (u16*)(ws + 16 * MB);   // 8 MB  [B][H][S][64]
    u16* Kh  = (u16*)(ws + 24 * MB);   // 8 MB  [B][H][S][64]
    u16* Vt  = (u16*)(ws + 32 * MB);   // 8 MB  [B][H][64][S]
    u16* ctx = (u16*)(ws + 40 * MB);   // 8 MB  [4096][1024]

    // fp32 -> bf16 (one time; keeps GEMM inner loops free of cvt VALU)
    mha_cvt_bf16<<<4096, 256, 0, stream>>>(x,  xb,  MTOT * D_);
    mha_cvt_bf16<<<1024, 256, 0, stream>>>(Wq, Wqb, D_ * D_);
    mha_cvt_bf16<<<1024, 256, 0, stream>>>(Wk, Wkb, D_ * D_);
    mha_cvt_bf16<<<1024, 256, 0, stream>>>(Wv, Wvb, D_ * D_);
    mha_cvt_bf16<<<1024, 256, 0, stream>>>(Wo, Wob, D_ * D_);

    dim3 ggrid(MTOT / 128, D_ / 128);
    mha_qkv_gemm<<<ggrid, 256, 0, stream>>>(xb, Wqb, bq, Qh, 0);
    mha_qkv_gemm<<<ggrid, 256, 0, stream>>>(xb, Wkb, bk, Kh, 0);
    mha_qkv_gemm<<<ggrid, 256, 0, stream>>>(xb, Wvb, bv, Vt, 1);

    mha_attn<<<dim3(2 * H_, S_ / 128), 128, 0, stream>>>(Qh, Kh, Vt, mask, ctx);

    mha_proj<<<ggrid, 256, 0, stream>>>(ctx, Wob, bo, (float*)d_out);
}
